// CSPNet_85598698209431
// MI455X (gfx1250) — matrix operand order repaired and checked by reading, not compiled, not run on hardware
//
#include <hip/hip_runtime.h>
#include <hip/hip_bf16.h>
#include <stdint.h>

// ---------------- problem constants ----------------
#define N_GRAPHS 128
#define ATOMS    24
#define NN       (N_GRAPHS * ATOMS)          // 3072 nodes
#define EE       (N_GRAPHS * ATOMS * ATOMS)  // 73728 edges
#define HH       512
#define LL       6
#define NFREQ    128
#define EXTW     800                          // 9 + 768 sin/cos + 23 zero pad
#define K_EDGE1  1824                         // 512 + 512 + 800  (57 * 32)

typedef __attribute__((ext_vector_type(16))) _Float16  v16h;
typedef __attribute__((ext_vector_type(8)))  float     v8f;
typedef __attribute__((ext_vector_type(4)))  uint32_t  u32x4;

union Frag16 { v16h h; u32x4 q[2]; uint32_t u[8]; };

// ---------------- GEMM tiling ----------------
#define BM 128
#define BN 128
#define BK 32
#define MT 4   // 16-row WMMA tiles per wave (wave covers 64 rows)
#define NT 2   // 16-col WMMA tiles per wave (wave covers 32 cols)

// Weights are pre-swizzled into WMMA B-fragment order:
//   tile t = kt*(512/16) + nt16  (kt: K/32 step, nt16: 16-col tile)
//   within tile: [lane 0..31][elem 0..15], value = W[kt*32 + (lane>>4)*16 + e][nt16*16 + (lane&15)]
// so one lane's whole fragment (16 f16 = 32B) is contiguous.

// MODE 0: edge GEMM1  A = gather(hn[src] | hn[dst] | eext), K=1824, out=silu->f16
// MODE 1: edge GEMM2  A = ef1_f16 (dense),                  K=512,  out=silu->f16
// MODE 2: node GEMM1  A = gather(hn | agg),                 K=1024, out=silu->f16
// MODE 3: node GEMM2  A = t_f16 (dense),                    K=512,  out: h_f32 += silu
// MODE 4: out  GEMM   A = hfin_f16 (dense),                 K=512,  out: f32 = acc+bias
template <int MODE>
__global__ __launch_bounds__(256) void gemm_kernel(
    const _Float16* __restrict__ A0, const _Float16* __restrict__ A1,
    const _Float16* __restrict__ A2,
    const int* __restrict__ srcIdx, const int* __restrict__ dstIdx,
    const _Float16* __restrict__ Bsw, const float* __restrict__ bias,
    _Float16* __restrict__ outH, float* __restrict__ outF, int K)
{
  __shared__ __align__(16) _Float16 As[BM][BK + 8];
  __shared__ __align__(16) _Float16 Bs[8 * 512];     // 8 fragment tiles, flat copy

  const int tid    = threadIdx.x;
  const int lane   = tid & 31;
  const int wave   = tid >> 5;
  const int waveM  = wave >> 2;       // 0..1
  const int waveN  = wave & 3;        // 0..3
  const int laneHi = lane >> 4;       // 0/1
  const int l16    = lane & 15;
  const int m0     = blockIdx.x * BM;
  const int n0     = blockIdx.y * BN;

  // each thread stages A rows r0 and r0+64 at column chunk colA
  const int r0   = tid >> 2;
  const int colA = (tid & 3) << 3;

  // gather indices loaded once (MODE 0 only)
  int gS0 = 0, gS1 = 0, gD0 = 0, gD1 = 0;
  if (MODE == 0) {
    gS0 = srcIdx[m0 + r0];
    gS1 = srcIdx[m0 + r0 + 64];
    gD0 = dstIdx[m0 + r0];
    gD1 = dstIdx[m0 + r0 + 64];
  }

  v8f acc[MT][NT] = {};

  for (int k0 = 0; k0 < K; k0 += BK) {
    // ---- stage A tile (128 x 32 f16); region select is uniform in k0 ----
    const _Float16 *pa0, *pa1;
    if (MODE == 0) {
      if (k0 < 512) {
        pa0 = A0 + (size_t)gS0 * HH + k0 + colA;
        pa1 = A0 + (size_t)gS1 * HH + k0 + colA;
      } else if (k0 < 1024) {
        pa0 = A1 + (size_t)gD0 * HH + (k0 - 512) + colA;
        pa1 = A1 + (size_t)gD1 * HH + (k0 - 512) + colA;
      } else {
        pa0 = A2 + (size_t)(m0 + r0) * EXTW + (k0 - 1024) + colA;
        pa1 = A2 + (size_t)(m0 + r0 + 64) * EXTW + (k0 - 1024) + colA;
      }
    } else if (MODE == 2) {
      if (k0 < 512) {
        pa0 = A0 + (size_t)(m0 + r0) * HH + k0 + colA;
        pa1 = A0 + (size_t)(m0 + r0 + 64) * HH + k0 + colA;
      } else {
        pa0 = A1 + (size_t)(m0 + r0) * HH + (k0 - 512) + colA;
        pa1 = A1 + (size_t)(m0 + r0 + 64) * HH + (k0 - 512) + colA;
      }
    } else {
      pa0 = A0 + (size_t)(m0 + r0) * K + k0 + colA;
      pa1 = A0 + (size_t)(m0 + r0 + 64) * K + k0 + colA;
    }
    *(u32x4*)&As[r0][colA]      = *(const u32x4*)pa0;
    *(u32x4*)&As[r0 + 64][colA] = *(const u32x4*)pa1;

    // ---- stage B fragments: 8 contiguous tiles = 8KB flat copy ----
    const _Float16* bsrc = Bsw + (((size_t)(k0 >> 5) * 32 + (n0 >> 4)) << 9);
    {
      const _Float16* bp = bsrc + tid * 16;
      *(u32x4*)&Bs[tid * 16]     = *(const u32x4*)bp;
      *(u32x4*)&Bs[tid * 16 + 8] = *(const u32x4*)(bp + 8);
    }
    // prefetch next K-step's B block (global_prefetch_b8)
    if (k0 + BK < K)
      __builtin_prefetch(bsrc + 16384 + tid * 32, 0, 0);
    __syncthreads();

    // ---- load fragments: all contiguous ds_load_b128 pairs ----
    Frag16 fa[MT], fb[NT];
#pragma unroll
    for (int mt = 0; mt < MT; ++mt) {
      const _Float16* ap = &As[waveM * 64 + mt * 16 + l16][laneHi * 16];
      fa[mt].q[0] = *(const u32x4*)ap;
      fa[mt].q[1] = *(const u32x4*)(ap + 8);
    }
#pragma unroll
    for (int nt = 0; nt < NT; ++nt) {
      const _Float16* bp = &Bs[((waveN * 2 + nt) * 32 + lane) * 16];
      fb[nt].q[0] = *(const u32x4*)bp;
      fb[nt].q[1] = *(const u32x4*)(bp + 8);
    }
#pragma unroll
    for (int mt = 0; mt < MT; ++mt)
#pragma unroll
      for (int nt = 0; nt < NT; ++nt)
        acc[mt][nt] = __builtin_amdgcn_wmma_f32_16x16x32_f16(
            false, fa[mt].h, false, fb[nt].h, (short)0, acc[mt][nt], false, false);
    __syncthreads();
  }

  // ---- epilogue (f32 C/D layout: lane n = lane&15, M = r + 8*(lane>>4)) ----
#pragma unroll
  for (int mt = 0; mt < MT; ++mt) {
#pragma unroll
    for (int nt = 0; nt < NT; ++nt) {
      int   colg = n0 + waveN * 32 + nt * 16 + l16;
      float bv   = bias[colg];
#pragma unroll
      for (int r = 0; r < 8; ++r) {
        int   rowg = m0 + waveM * 64 + mt * 16 + (laneHi << 3) + r;
        float x    = acc[mt][nt][r] + bv;
        if (MODE != 4) x = x / (1.0f + __expf(-x));       // silu
        size_t o = (size_t)rowg * HH + colg;
        if (MODE == 3)      outF[o] += x;                 // residual h += silu(...)
        else if (MODE == 4) outF[o] = x;                  // final: acc + bias
        else                outH[o] = (_Float16)x;
      }
    }
  }
}

// ---------------- small helper kernels ----------------
__global__ void latip_kernel(const float* __restrict__ lat, float* __restrict__ latip) {
  int g = blockIdx.x, t = threadIdx.x;
  if (t < 9) {
    int i = t / 3, k = t % 3;
    const float* L = lat + g * 9;
    latip[g * 9 + t] = L[i * 3 + 0] * L[k * 3 + 0] + L[i * 3 + 1] * L[k * 3 + 1] +
                       L[i * 3 + 2] * L[k * 3 + 2];
  }
}

// per-edge static features: [lat_ip(9) | sin(384) | cos(384) | pad(23)] -> f16
__global__ void eext_kernel(const int* __restrict__ ei, const int* __restrict__ e2g,
                            const float* __restrict__ frac, const float* __restrict__ latip,
                            _Float16* __restrict__ eext) {
  int e = blockIdx.x;
  int s = ei[e], d = ei[EE + e], g = e2g[e];
  float fd[3];
#pragma unroll
  for (int k = 0; k < 3; ++k) {
    float v = frac[d * 3 + k] - frac[s * 3 + k];
    fd[k] = v - floorf(v);                     // python float %
  }
  for (int c = threadIdx.x; c < EXTW; c += blockDim.x) {
    float v;
    if (c < 9) {
      v = latip[g * 9 + c];
    } else if (c < 777) {
      int q = c - 9;
      int iscos = q >= 384;
      int r = iscos ? q - 384 : q;
      int dim = r / NFREQ, f = r % NFREQ;
      float ang = fd[dim] * (6.283185307179586f * (float)f);
      v = iscos ? __cosf(ang) : __sinf(ang);
    } else {
      v = 0.0f;
    }
    eext[(size_t)e * EXTW + c] = (_Float16)v;
  }
}

__global__ void embed_kernel(const int* __restrict__ at, const float* __restrict__ emb,
                             float* __restrict__ h) {
  int n = blockIdx.x, t = threadIdx.x;
  int a = at[n];
  h[(size_t)n * HH + t]       = emb[(size_t)a * HH + t];
  h[(size_t)n * HH + t + 256] = emb[(size_t)a * HH + t + 256];
}

__global__ void ln_kernel(const float* __restrict__ h, const float* __restrict__ w,
                          const float* __restrict__ b, _Float16* __restrict__ outh) {
  __shared__ float s[256];
  int n = blockIdx.x, t = threadIdx.x;
  float x0 = h[(size_t)n * HH + t];
  float x1 = h[(size_t)n * HH + t + 256];
  s[t] = x0 + x1;
  __syncthreads();
  for (int o = 128; o > 0; o >>= 1) { if (t < o) s[t] += s[t + o]; __syncthreads(); }
  float mu = s[0] * (1.0f / HH);
  __syncthreads();
  float d0 = x0 - mu, d1 = x1 - mu;
  s[t] = d0 * d0 + d1 * d1;
  __syncthreads();
  for (int o = 128; o > 0; o >>= 1) { if (t < o) s[t] += s[t + o]; __syncthreads(); }
  float rs = rsqrtf(s[0] * (1.0f / HH) + 1e-5f);
  outh[(size_t)n * HH + t]       = (_Float16)(d0 * rs * w[t] + b[t]);
  outh[(size_t)n * HH + t + 256] = (_Float16)(d1 * rs * w[t + 256] + b[t + 256]);
}

// scatter_mean over src: edges are src-major, each node owns 24 contiguous rows
__global__ void agg_kernel(const _Float16* __restrict__ ef, _Float16* __restrict__ agg) {
  int n = blockIdx.x, t = threadIdx.x;
#pragma unroll
  for (int i = 0; i < 2; ++i) {
    int c = t + i * 256;
    float sum = 0.0f;
#pragma unroll
    for (int j = 0; j < ATOMS; ++j)
      sum += (float)ef[(size_t)(n * ATOMS + j) * HH + c];
    agg[(size_t)n * HH + c] = (_Float16)(sum * (1.0f / ATOMS));
  }
}

// convert f32 weights [layers][Ksrc][512] -> f16 swizzled [layers][K/32][32 tiles][32 lanes][16]
__global__ void cvt_sw_kernel(const float* __restrict__ src, _Float16* __restrict__ dst,
                              int K, int Ksrc, int layers) {
  long i   = (long)blockIdx.x * 256 + threadIdx.x;
  long per = (long)K * HH;
  if (i >= per * layers) return;
  int  l    = (int)(i / per);
  long j    = i - (long)l * per;
  int  e    = (int)(j & 15);
  int  r    = (int)((j >> 4) & 31);
  int  tile = (int)(j >> 9);
  int  kt   = tile >> 5;
  int  nt16 = tile & 31;
  int  k    = (kt << 5) + ((r >> 4) << 4) + e;
  int  n    = (nt16 << 4) + (r & 15);
  dst[i] = (k < Ksrc) ? (_Float16)src[((size_t)l * Ksrc + k) * HH + n] : (_Float16)0.0f;
}

// ---------------- launch ----------------
extern "C" void kernel_launch(void* const* d_in, const int* in_sizes, int n_in,
                              void* d_out, int out_size, void* d_ws, size_t ws_size,
                              hipStream_t stream) {
  (void)in_sizes; (void)n_in; (void)out_size; (void)ws_size;
  const int*   atom_types = (const int*)d_in[0];
  const int*   edge_index = (const int*)d_in[1];
  const int*   e2g        = (const int*)d_in[2];
  const float* frac       = (const float*)d_in[3];
  const float* lat        = (const float*)d_in[4];
  const float* emb        = (const float*)d_in[5];
  const float* ln_w       = (const float*)d_in[6];
  const float* ln_b       = (const float*)d_in[7];
  const float* e_w1       = (const float*)d_in[8];
  const float* e_b1       = (const float*)d_in[9];
  const float* e_w2       = (const float*)d_in[10];
  const float* e_b2       = (const float*)d_in[11];
  const float* n_w1       = (const float*)d_in[12];
  const float* n_b1       = (const float*)d_in[13];
  const float* n_w2       = (const float*)d_in[14];
  const float* n_b2       = (const float*)d_in[15];
  const float* fln_w      = (const float*)d_in[16];
  const float* fln_b      = (const float*)d_in[17];
  const float* out_w      = (const float*)d_in[18];
  const float* out_b      = (const float*)d_in[19];

  // carve workspace
  char* w = (char*)d_ws;
  auto carve = [&](size_t bytes) { void* p = (void*)w; w += (bytes + 255) & ~(size_t)255; return p; };
  _Float16* eext   = (_Float16*)carve((size_t)EE * EXTW * 2);
  _Float16* ef1    = (_Float16*)carve((size_t)EE * HH * 2);
  _Float16* ef     = (_Float16*)carve((size_t)EE * HH * 2);
  _Float16* hnf16  = (_Float16*)carve((size_t)NN * HH * 2);
  _Float16* aggf16 = (_Float16*)carve((size_t)NN * HH * 2);
  _Float16* tf16   = (_Float16*)carve((size_t)NN * HH * 2);
  _Float16* w1f    = (_Float16*)carve((size_t)LL * K_EDGE1 * HH * 2);
  _Float16* w2f    = (_Float16*)carve((size_t)LL * HH * HH * 2);
  _Float16* nw1f   = (_Float16*)carve((size_t)LL * 2 * HH * HH * 2);
  _Float16* nw2f   = (_Float16*)carve((size_t)LL * HH * HH * 2);
  _Float16* outwf  = (_Float16*)carve((size_t)HH * HH * 2);
  float*    latip  = (float*)carve((size_t)N_GRAPHS * 9 * 4);
  float*    hbuf   = (float*)carve((size_t)NN * HH * 4);

  const int* srcI = edge_index;
  const int* dstI = edge_index + EE;

  // ---- one-time precompute ----
  latip_kernel<<<N_GRAPHS, 16, 0, stream>>>(lat, latip);
  {
    long tot = (long)LL * K_EDGE1 * HH;
    cvt_sw_kernel<<<(int)((tot + 255) / 256), 256, 0, stream>>>(e_w1, w1f, K_EDGE1, 1801, LL);
  }
  {
    long tot = (long)LL * HH * HH;
    cvt_sw_kernel<<<(int)((tot + 255) / 256), 256, 0, stream>>>(e_w2, w2f, HH, HH, LL);
    cvt_sw_kernel<<<(int)((tot + 255) / 256), 256, 0, stream>>>(n_w2, nw2f, HH, HH, LL);
  }
  {
    long tot = (long)LL * 2 * HH * HH;
    cvt_sw_kernel<<<(int)((tot + 255) / 256), 256, 0, stream>>>(n_w1, nw1f, 2 * HH, 2 * HH, LL);
  }
  cvt_sw_kernel<<<(HH * HH + 255) / 256, 256, 0, stream>>>(out_w, outwf, HH, HH, 1);
  eext_kernel<<<EE, 256, 0, stream>>>(edge_index, e2g, frac, latip, eext);
  embed_kernel<<<NN, 256, 0, stream>>>(atom_types, emb, hbuf);

  dim3 gridE(EE / BM, HH / BN);   // 576 x 4
  dim3 gridN(NN / BM, HH / BN);   // 24 x 4

  for (int l = 0; l < LL; ++l) {
    ln_kernel<<<NN, 256, 0, stream>>>(hbuf, ln_w + l * HH, ln_b + l * HH, hnf16);
    gemm_kernel<0><<<gridE, 256, 0, stream>>>(hnf16, hnf16, eext, srcI, dstI,
        w1f + (size_t)l * K_EDGE1 * HH, e_b1 + l * HH, ef1, nullptr, K_EDGE1);
    gemm_kernel<1><<<gridE, 256, 0, stream>>>(ef1, nullptr, nullptr, nullptr, nullptr,
        w2f + (size_t)l * HH * HH, e_b2 + l * HH, ef, nullptr, HH);
    agg_kernel<<<NN, 256, 0, stream>>>(ef, aggf16);
    gemm_kernel<2><<<gridN, 256, 0, stream>>>(hnf16, aggf16, nullptr, nullptr, nullptr,
        nw1f + (size_t)l * 2 * HH * HH, n_b1 + l * HH, tf16, nullptr, 2 * HH);
    gemm_kernel<3><<<gridN, 256, 0, stream>>>(tf16, nullptr, nullptr, nullptr, nullptr,
        nw2f + (size_t)l * HH * HH, n_b2 + l * HH, nullptr, hbuf, HH);
  }

  ln_kernel<<<NN, 256, 0, stream>>>(hbuf, fln_w, fln_b, hnf16);
  gemm_kernel<4><<<gridN, 256, 0, stream>>>(hnf16, nullptr, nullptr, nullptr, nullptr,
      outwf, out_b, nullptr, (float*)d_out, HH);
}
